// FilterbankFeatures_42434276884667
// MI455X (gfx1250) — matrix operand order repaired and checked by reading, not compiled, not run on hardware
//
#include <hip/hip_runtime.h>
#include <hip/hip_bf16.h>
#include <math.h>

// ---- problem constants (from reference) ----
#define HOP          160
#define WIN_LEN      320
#define WPAD         96          // (512-320)/2
#define NFFT         512
#define NBINS        257
#define NMELS        64
#define NB           16
#define TLEN         240000
#define NFRAMES      1501        // 1 + T/HOP
#define NFRAMES_OUT  1500        // int(max(seq)) since seq_len[0] == T
#define KPAD         288         // 9 * 32  (K padded for WMMA)
#define NPAD         1504        // 94 * 16 (frames padded for WMMA)
#define KCHUNKS      9
#define NTILES_N     94
#define NTILES_M     4
#define DITHER2      1e-10f
#define CONSTV       1e-5f

typedef __attribute__((ext_vector_type(16))) __bf16 bf16x16;
typedef __attribute__((ext_vector_type(8)))  float  f32x8;

// ---------------- zero-fill padded power buffer ----------------
__global__ void zero_kernel(float* __restrict__ p, int n) {
    int i = blockIdx.x * blockDim.x + threadIdx.x;
    if (i < n) p[i] = 0.0f;
}

// ---------------- preemph + frame + window + 512pt FFT -> power ----------------
// one 256-thread workgroup per (frame t, batch b); radix-2 DIT in LDS
__global__ __launch_bounds__(256) void fft_power_kernel(const float* __restrict__ x,
                                                        const float* __restrict__ window,
                                                        float* __restrict__ P) {
    __shared__ float sRe[NFFT];
    __shared__ float sIm[NFFT];
    const int t   = blockIdx.x;          // 0..1500
    const int b   = blockIdx.y;          // 0..15
    const int tid = threadIdx.x;         // 0..255
    const float* xb = x + (size_t)b * TLEN;

    // load two samples per thread, bit-reversed into LDS
    for (int jj = 0; jj < 2; ++jj) {
        int j = tid + jj * 256;                       // 0..511 within frame
        long q = (long)t * HOP + j - (NFFT / 2);      // signal index before reflect
        int r;
        if (q < 0)            r = (int)(-q);                         // reflect left
        else if (q >= TLEN)   r = (int)(2L * TLEN - 2 - q);          // reflect right
        else                  r = (int)q;
        float y = (r == 0) ? xb[0] : (xb[r] - 0.97f * xb[r - 1]);    // preemphasis
        float w = (j >= WPAD && j < WPAD + WIN_LEN) ? window[j - WPAD] : 0.0f;
        int rev = (int)(__brev((unsigned)j) >> 23);   // 9-bit bit reversal
        sRe[rev] = y * w;
        sIm[rev] = 0.0f;
    }
    __syncthreads();

    // 9 radix-2 stages, 256 butterflies/stage
    #pragma unroll
    for (int s = 1; s <= 9; ++s) {
        int half = 1 << (s - 1);
        int blk  = tid >> (s - 1);
        int pos  = tid & (half - 1);
        int i0   = (blk << s) + pos;
        int i1   = i0 + half;
        float ang = -3.14159265358979323846f * (float)pos / (float)half;
        float wi, wr;
        __sincosf(ang, &wi, &wr);                     // w = e^{-i*theta}
        float re1 = sRe[i1], im1 = sIm[i1];
        float tr = wr * re1 - wi * im1;
        float ti = wr * im1 + wi * re1;
        float re0 = sRe[i0], im0 = sIm[i0];
        sRe[i1] = re0 - tr;  sIm[i1] = im0 - ti;
        sRe[i0] = re0 + tr;  sIm[i0] = im0 + ti;
        __syncthreads();
    }

    // power bins 0..256 (+ dither^2), stored transposed P[b][k][t]
    float pw = sRe[tid] * sRe[tid] + sIm[tid] * sIm[tid] + DITHER2;
    P[((size_t)b * KPAD + tid) * NPAD + t] = pw;
    if (tid == 0) {
        float p2 = sRe[256] * sRe[256] + sIm[256] * sIm[256] + DITHER2;
        P[((size_t)b * KPAD + 256) * NPAD + t] = p2;
    }
}

// ---------------- pack fb into WMMA A-layout (bf16 hi/lo), zero-padded ----------------
// element i = ((mt*KCHUNKS + kc)*32 + lane)*16 + e
__global__ __launch_bounds__(256) void pack_fb_kernel(const float* __restrict__ fb,
                                                      __bf16* __restrict__ hi,
                                                      __bf16* __restrict__ lo) {
    int i = blockIdx.x * blockDim.x + threadIdx.x;
    if (i >= NTILES_M * KCHUNKS * 32 * 16) return;
    int e    = i & 15;
    int lane = (i >> 4) & 31;
    int mtkc = i >> 9;
    int kc   = mtkc % KCHUNKS;
    int mt   = mtkc / KCHUNKS;
    int m    = mt * 16 + (lane & 15);
    int koff = ((lane >> 4) ? 8 : 0) + ((e < 8) ? e : (16 + (e - 8)));
    int k    = kc * 32 + koff;
    float v  = (k < NBINS) ? fb[m * NBINS + k] : 0.0f;
    __bf16 vh = (__bf16)v;
    hi[i] = vh;
    lo[i] = (__bf16)(v - (float)vh);
}

// ---------------- mel GEMM: async-to-LDS B staging + split-bf16 WMMA ----------------
// 4 waves per block, one M-tile each, sharing the 32x16 B tile staged via
// global_load_async_to_lds_b128 (ASYNCcnt path)
__global__ __launch_bounds__(128) void mel_wmma_kernel(const __bf16* __restrict__ fbAhi,
                                                       const __bf16* __restrict__ fbAlo,
                                                       const float* __restrict__ P,
                                                       float* __restrict__ LM) {
    __shared__ float btile[32 * 16];      // K x N tile of power spectrum (2 KB)
    const int tid  = threadIdx.x;         // 0..127
    const int lane = tid & 31;            // wave32 lane
    const int mt   = tid >> 5;            // one M-tile per wave
    const int nt   = blockIdx.x;          // 0..93
    const int b    = blockIdx.y;          // 0..15
    const int hh   = lane >> 4;
    const int l16  = lane & 15;

    const float* Pb = P + (size_t)b * KPAD * NPAD;

    // async-copy assignment: 128 threads x 16B = whole 2KB tile
    const int row = tid >> 2;             // 0..31 (K within chunk)
    const int c4  = (tid & 3) << 2;       // 0,4,8,12 (N within tile, x4 floats)
    const unsigned lds_off = (unsigned)(size_t)(void*)&btile[row * 16 + c4];

    f32x8 acc = {0.f, 0.f, 0.f, 0.f, 0.f, 0.f, 0.f, 0.f};

    for (int kc = 0; kc < KCHUNKS; ++kc) {
        __syncthreads();                  // prior-iteration LDS reads are done
        unsigned goff = (unsigned)((((kc * 32 + row) * NPAD) + nt * 16 + c4) * 4);
        asm volatile("global_load_async_to_lds_b128 %0, %1, %2"
                     :: "v"(lds_off), "v"(goff), "s"(Pb)
                     : "memory");
        asm volatile("s_wait_asynccnt 0x0" ::: "memory");
        __syncthreads();                  // full tile visible to all 4 waves

        // A tile: pre-packed, contiguous 32B vector loads per lane
        const size_t abase = (((size_t)(mt * KCHUNKS + kc) * 32) + lane) << 4;
        bf16x16 a_hi = *(const bf16x16*)(fbAhi + abase);
        bf16x16 a_lo = *(const bf16x16*)(fbAlo + abase);

        // B tile from LDS, matching A packing: k = hh*8+e (e<8), 16+hh*8+(e-8) (e>=8), n = l16
        bf16x16 b_hi, b_lo;
        #pragma unroll
        for (int e = 0; e < 16; ++e) {
            float v = btile[(hh * 8 + ((e < 8) ? e : (16 + (e - 8)))) * 16 + l16];
            __bf16 vh = (__bf16)v;
            b_hi[e] = vh;
            b_lo[e] = (__bf16)(v - (float)vh);
        }
        // split-precision accumulation (~2^-17 relative error vs f32)
        acc = __builtin_amdgcn_wmma_f32_16x16x32_bf16(false, a_hi, false, b_hi,
                                                      (short)0, acc, false, false);
        acc = __builtin_amdgcn_wmma_f32_16x16x32_bf16(false, a_hi, false, b_lo,
                                                      (short)0, acc, false, false);
        acc = __builtin_amdgcn_wmma_f32_16x16x32_bf16(false, a_lo, false, b_hi,
                                                      (short)0, acc, false, false);
    }

    // D layout: VGPR r -> M = r + hh*8, N = l16 ; write log-mel
    const int t = nt * 16 + l16;
    if (t < NFRAMES) {
        #pragma unroll
        for (int r = 0; r < 8; ++r) {
            int mm = mt * 16 + hh * 8 + r;
            LM[((size_t)(b * NMELS + mm)) * NPAD + t] = logf(acc[r] + 1e-20f);
        }
    }
}

// ---------------- masked mean / std per (batch, mel) ----------------
__global__ __launch_bounds__(256) void stats_kernel(const float* __restrict__ LM,
                                                    const int* __restrict__ seq_len,
                                                    float* __restrict__ mean,
                                                    float* __restrict__ stdv) {
    const int bm = blockIdx.x;           // 0..1023
    const int b  = bm / NMELS;
    const int nseq = (seq_len[b] + HOP - 1) / HOP;
    const float* row = LM + (size_t)bm * NPAD;

    float s = 0.f, ss = 0.f;
    for (int t = threadIdx.x; t < nseq; t += 256) {
        float v = row[t];
        s += v; ss += v * v;
    }
    __shared__ float sh1[256], sh2[256];
    sh1[threadIdx.x] = s; sh2[threadIdx.x] = ss;
    __syncthreads();
    for (int o = 128; o > 0; o >>= 1) {
        if (threadIdx.x < o) {
            sh1[threadIdx.x] += sh1[threadIdx.x + o];
            sh2[threadIdx.x] += sh2[threadIdx.x + o];
        }
        __syncthreads();
    }
    if (threadIdx.x == 0) {
        float fn  = (float)nseq;
        float mu  = sh1[0] / fn;
        float var = (sh2[0] - fn * mu * mu) / (fn - 1.0f);
        float sd  = sqrtf(var) + CONSTV * (float)(NB - b);
        mean[bm] = mu;
        stdv[bm] = sd;
    }
}

// ---------------- normalize + truncate to 1500 frames ----------------
__global__ void norm_kernel(const float* __restrict__ LM,
                            const float* __restrict__ mean,
                            const float* __restrict__ stdv,
                            float* __restrict__ out) {
    int i = blockIdx.x * blockDim.x + threadIdx.x;
    if (i >= NB * NMELS * NFRAMES_OUT) return;
    int t  = i % NFRAMES_OUT;
    int bm = i / NFRAMES_OUT;
    out[i] = (LM[(size_t)bm * NPAD + t] - mean[bm]) / stdv[bm];
}

extern "C" void kernel_launch(void* const* d_in, const int* in_sizes, int n_in,
                              void* d_out, int out_size, void* d_ws, size_t ws_size,
                              hipStream_t stream) {
    const float* x       = (const float*)d_in[0];   // (16, 240000) f32
    const int*   seq_len = (const int*)d_in[1];     // (16,) int
    const float* fb      = (const float*)d_in[2];   // (64, 257) f32
    const float* window  = (const float*)d_in[3];   // (320,) f32
    float* out = (float*)d_out;                     // (16, 64, 1500) f32

    // workspace carve-up (~34 MB)
    float*  P     = (float*)d_ws;                             // 16*288*1504
    float*  LM    = P    + (size_t)NB * KPAD * NPAD;          // 16*64*1504
    float*  mean  = LM   + (size_t)NB * NMELS * NPAD;         // 1024
    float*  stdv  = mean + NB * NMELS;                        // 1024
    __bf16* fbAhi = (__bf16*)(stdv + NB * NMELS);             // 4*9*32*16 bf16
    __bf16* fbAlo = fbAhi + (size_t)NTILES_M * KCHUNKS * 32 * 16;

    const int nP = NB * KPAD * NPAD;
    zero_kernel<<<(nP + 255) / 256, 256, 0, stream>>>(P, nP);

    const int nFb = NTILES_M * KCHUNKS * 32 * 16;
    pack_fb_kernel<<<(nFb + 255) / 256, 256, 0, stream>>>(fb, fbAhi, fbAlo);

    fft_power_kernel<<<dim3(NFRAMES, NB), 256, 0, stream>>>(x, window, P);

    mel_wmma_kernel<<<dim3(NTILES_N, NB), 128, 0, stream>>>(fbAhi, fbAlo, P, LM);

    stats_kernel<<<NB * NMELS, 256, 0, stream>>>(LM, seq_len, mean, stdv);

    const int nOut = NB * NMELS * NFRAMES_OUT;
    norm_kernel<<<(nOut + 255) / 256, 256, 0, stream>>>(LM, mean, stdv, out);
}
